// Cosais_68908455297199
// MI455X (gfx1250) — matrix operand order repaired and checked by reading, not compiled, not run on hardware
//
#include <hip/hip_runtime.h>
#include <hip/hip_bf16.h>

// ---------------------------------------------------------------------------
// Types for CDNA5 WMMA (wave32)
// ---------------------------------------------------------------------------
typedef __bf16 bf16_t;
typedef __attribute__((ext_vector_type(16))) __bf16 v16bf;
typedef __attribute__((ext_vector_type(8)))  float  v8f;
typedef __attribute__((ext_vector_type(4)))  int    v4i;

#define ACT_NONE 0
#define ACT_GELU 1

#if defined(__has_builtin)
#  if __has_builtin(__builtin_amdgcn_global_load_async_to_lds_b128)
#    define HAVE_ASYNC_LDS 1
#  endif
#endif

static __device__ __forceinline__ bf16_t f2b(float f) {
    unsigned u = __float_as_uint(f);
    unsigned r = (u + 0x7FFFu + ((u >> 16) & 1u)) >> 16;   // round-to-nearest-even
    unsigned short s = (unsigned short)r;
    return __builtin_bit_cast(bf16_t, s);
}

static __device__ __forceinline__ v16bf bz16() {
    union { v16bf v; unsigned u[8]; } z;
#pragma unroll
    for (int i = 0; i < 8; ++i) z.u[i] = 0u;
    return z.v;
}

// ---------------------------------------------------------------------------
// 16-byte global -> LDS stage. On gfx1250 uses the async copy path
// (GLOBAL_LOAD_ASYNC_TO_LDS_B128, ASYNCcnt); falls back to load + ds_store.
// Builtin params are int4*-typed; clang rewrites pointer address spaces.
// ---------------------------------------------------------------------------
static __device__ __forceinline__ void stage16(const bf16_t* g, bf16_t* l) {
#ifdef HAVE_ASYNC_LDS
    __builtin_amdgcn_global_load_async_to_lds_b128((v4i*)g, (v4i*)l, 0, 0);
#else
    *reinterpret_cast<uint4*>(l) = *reinterpret_cast<const uint4*>(g);
#endif
}

static __device__ __forceinline__ void wait_async_copies() {
#ifdef HAVE_ASYNC_LDS
#  if __has_builtin(__builtin_amdgcn_s_wait_asynccnt)
    __builtin_amdgcn_s_wait_asynccnt(0);
#  else
    asm volatile("s_wait_asynccnt 0x0" ::: "memory");
#  endif
#endif
}

// ---------------------------------------------------------------------------
// A-fragment loader: 16x32 bf16, row-major source, stride lda (elements).
// Contiguous k order (k = 16*half + s) per lane, matching the B fragments:
//   dwords [8*half .. 8*half+7]  ->  2x b128 loads.
// ---------------------------------------------------------------------------
static __device__ __forceinline__ v16bf load_a32(const bf16_t* __restrict__ A,
                                                 int lda, int row0, int k0, int lane) {
    const int half = lane >> 4;
    const int r    = lane & 15;
    const uint4* p = reinterpret_cast<const uint4*>(A + (size_t)(row0 + r) * lda + k0);
    union { v16bf v; uint4 q[2]; } a;
    a.q[0] = p[2 * half];
    a.q[1] = p[2 * half + 1];
    return a.v;
}

// ---------------------------------------------------------------------------
// f32 -> bf16 WITH transpose:  in[K,N] (per layer) -> out[N,K].
// 32x32 LDS tiles; grid.z = layer, strides K*N.
// ---------------------------------------------------------------------------
__global__ __launch_bounds__(256) void f2bf_t_kernel(const float* __restrict__ in,
                                                     bf16_t* __restrict__ out,
                                                     int K, int N) {
    __shared__ float tile[32][33];
    const size_t lo = (size_t)blockIdx.z * K * N;
    const int k0 = blockIdx.y * 32, n0 = blockIdx.x * 32;
    for (int i = threadIdx.y; i < 32; i += 8) {
        int k = k0 + i, n = n0 + threadIdx.x;
        tile[i][threadIdx.x] = (k < K && n < N) ? in[lo + (size_t)k * N + n] : 0.f;
    }
    __syncthreads();
    for (int i = threadIdx.y; i < 32; i += 8) {
        int n = n0 + i, k = k0 + threadIdx.x;
        if (n < N && k < K) out[lo + (size_t)n * K + k] = f2b(tile[threadIdx.x][i]);
    }
}

// ---------------------------------------------------------------------------
// Four-hot embedding + positional embedding -> f32 residual stream
// ---------------------------------------------------------------------------
__global__ void embed_kernel(const int* __restrict__ idxs,
                             const float* __restrict__ lat, const float* __restrict__ lon,
                             const float* __restrict__ sog, const float* __restrict__ cog,
                             const float* __restrict__ pos, float* __restrict__ x) {
    const int D = 768, T = 1024;
    int i = blockIdx.x * blockDim.x + threadIdx.x;
    if (i >= 4096 * D) return;
    int row = i / D, d = i - row * D;
    int t = row & (T - 1);
    const int* id = idxs + (size_t)row * 4;
    float v;
    if (d < 256)        v = lat[id[0] * 256 + d];
    else if (d < 512)   v = lon[id[1] * 256 + (d - 256)];
    else if (d < 640)   v = sog[id[2] * 128 + (d - 512)];
    else                v = cog[id[3] * 128 + (d - 640)];
    x[i] = v + pos[(size_t)t * D + d];
}

// ---------------------------------------------------------------------------
// LayerNorm (one block per row), writes bf16 for the following WMMA GEMM
// ---------------------------------------------------------------------------
__global__ __launch_bounds__(256) void ln_kernel(const float* __restrict__ x,
                                                 const float* __restrict__ g,
                                                 const float* __restrict__ b,
                                                 bf16_t* __restrict__ out) {
    const int D = 768;
    int row = blockIdx.x;
    const float* xr = x + (size_t)row * D;
    __shared__ float red[256];

    float s = 0.f;
    for (int d = threadIdx.x; d < D; d += 256) s += xr[d];
    red[threadIdx.x] = s; __syncthreads();
    for (int o = 128; o > 0; o >>= 1) {
        if (threadIdx.x < o) red[threadIdx.x] += red[threadIdx.x + o];
        __syncthreads();
    }
    float mean = red[0] * (1.f / D);
    __syncthreads();

    float vs = 0.f;
    for (int d = threadIdx.x; d < D; d += 256) { float t = xr[d] - mean; vs += t * t; }
    red[threadIdx.x] = vs; __syncthreads();
    for (int o = 128; o > 0; o >>= 1) {
        if (threadIdx.x < o) red[threadIdx.x] += red[threadIdx.x + o];
        __syncthreads();
    }
    float rstd = rsqrtf(red[0] * (1.f / D) + 1e-5f);

    for (int d = threadIdx.x; d < D; d += 256)
        out[(size_t)row * D + d] = f2b((xr[d] - mean) * rstd * g[d] + b[d]);
}

// ---------------------------------------------------------------------------
// WMMA bf16 GEMM:  C[M,N] = act(A[M,K] @ B[K,N] + bias) (+resid)
// BT is the transposed weight [N,K]. Block = 256 threads (8 waves); each wave
// computes a 32x64 strip; block tile = 256 x 64.
// The shared 64x32 B tile is staged into LDS with async global->LDS copies
// (one GLOBAL_LOAD_ASYNC_TO_LDS_B128 per wave per K-chunk), double-buffered:
//   s_wait_asynccnt 0 ; barrier ; issue chunk i+1 -> buf^1 ; compute buf.
// B fragments then come from LDS as contiguous 32B ds_load_b128 pairs.
// M multiple of 256, K multiple of 32; N edge handled by source-col clamp +
// guarded stores. WBT: store bf16 transposed into [B,H,DH,T] (used for V).
// ---------------------------------------------------------------------------
template<int ACT, bool RESID, bool WF32, bool WBF16, bool WBT>
__global__ __launch_bounds__(256) void gemm_bf16_kernel(
        const bf16_t* __restrict__ A, const bf16_t* __restrict__ BT,
        const float* __restrict__ bias, const float* __restrict__ resid,
        float* __restrict__ Cf, bf16_t* __restrict__ Cb,
        int M, int N, int K) {
    __shared__ __align__(32) bf16_t smB[2][64 * 32];   // 2 x 4KB double buffer

    const int tid  = threadIdx.x;
    const int lane = tid & 31;
    const int wave = tid >> 5;
    const int row0 = blockIdx.x * 256 + wave * 32;
    const int col0 = blockIdx.y * 64;
    const int half = lane >> 4, cc = lane & 15;

    // ---- async copy mapping: thread -> (column, 16B K-segment) ----
    const int ccol = tid >> 2;            // 0..63
    const int ck   = (tid & 3) * 8;       // 0,8,16,24 (elements)
    int gcol = col0 + ccol;
    if (gcol >= N) gcol = N - 1;          // clamp: no OOB reads, stores guarded
    const bf16_t* gsrc = BT + (size_t)gcol * K + ck;
    bf16_t* ldst0 = &smB[0][ccol * 32 + ck];
    bf16_t* ldst1 = &smB[1][ccol * 32 + ck];

    v8f z = {0.f, 0.f, 0.f, 0.f, 0.f, 0.f, 0.f, 0.f};
    v8f acc[2][4];
#pragma unroll
    for (int rt = 0; rt < 2; ++rt)
#pragma unroll
        for (int t = 0; t < 4; ++t) acc[rt][t] = z;

    const int nk = K >> 5;
    stage16(gsrc, ldst0);                 // chunk 0 -> buf 0

    for (int i = 0; i < nk; ++i) {
        const int k0 = i << 5;
        wait_async_copies();              // own async stores to LDS landed
        __syncthreads();                  // all waves' tile slices visible

        if (i + 1 < nk)
            stage16(gsrc + k0 + 32, (i & 1) ? ldst0 : ldst1);

        if (k0 + 64 < K)                  // CDNA5 global_prefetch_b8 on A
            __builtin_prefetch(A + (size_t)(row0 + cc) * K + k0 + 64, 0, 0);

        v16bf a0 = load_a32(A, K, row0,      k0, lane);
        v16bf a1 = load_a32(A, K, row0 + 16, k0, lane);
        const bf16_t* bb = smB[i & 1];
#pragma unroll
        for (int t = 0; t < 4; ++t) {
            const v16bf* bp = reinterpret_cast<const v16bf*>(
                bb + (16 * t + cc) * 32 + 16 * half);
            v16bf b = *bp;
            acc[0][t] = __builtin_amdgcn_wmma_f32_16x16x32_bf16(
                            false, a0, false, b, (short)0, acc[0][t], false, false);
            acc[1][t] = __builtin_amdgcn_wmma_f32_16x16x32_bf16(
                            false, a1, false, b, (short)0, acc[1][t], false, false);
        }
        // WAR on this buffer: it is rewritten only after the next iteration's barrier: no extra sync.
    }

#pragma unroll
    for (int rt = 0; rt < 2; ++rt) {
#pragma unroll
        for (int t = 0; t < 4; ++t) {
            int col = col0 + 16 * t + cc;
            if (col >= N) continue;
#pragma unroll
            for (int j = 0; j < 8; ++j) {
                int row = row0 + 16 * rt + j + 8 * half;
                float v = acc[rt][t][j] + (bias ? bias[col] : 0.f);
                if constexpr (ACT == ACT_GELU)
                    v = 0.5f * v * (1.f + erff(v * 0.70710678118654752f));
                size_t o = (size_t)row * N + col;
                if constexpr (RESID) v += resid[o];
                if constexpr (WF32)  Cf[o] = v;
                if constexpr (WBF16) Cb[o] = f2b(v);
                if constexpr (WBT) {
                    // [B,H,DH,T]: index = (b*768 + col)*1024 + t  (N==768, T==1024)
                    int bb2 = row >> 10;
                    Cb[((size_t)bb2 * 768 + col) * 1024 + (row & 1023)] = f2b(v);
                }
            }
        }
    }
}

// ---------------------------------------------------------------------------
// Flash attention (causal), WMMA for both S = Q K^T and O = P V.
// Grid: (B*H, T/64). Block = 128 threads (4 waves); each wave owns 16 queries.
// K^T fragments: contiguous 32B loads from row-major K.
// V  fragments: contiguous 32B loads from transposed V [B,H,DH,T].
// Online softmax in registers; P staged through per-wave LDS tile to convert
// C-layout (f32) -> A-layout (bf16); reloaded with ds_load_b128.
// ---------------------------------------------------------------------------
__global__ __launch_bounds__(128) void attn_kernel(
        const bf16_t* __restrict__ qbf, const bf16_t* __restrict__ kbf,
        const bf16_t* __restrict__ vt, bf16_t* __restrict__ ybf) {
    const int T = 1024, Dm = 768, DH = 64, Hh = 12;
    const int bh = blockIdx.x;
    const int b = bh / Hh, h = bh % Hh;
    const int qt = blockIdx.y;
    const int lane = threadIdx.x & 31;
    const int wave = threadIdx.x >> 5;
    const int half = lane >> 4, cc = lane & 15;

    __shared__ __align__(32) bf16_t lds_p[4 * 16 * 64];
    bf16_t* myp = lds_p + wave * 16 * 64;

    const bf16_t* qb  = qbf + (size_t)b * T * Dm + h * DH;
    const bf16_t* kb  = kbf + (size_t)b * T * Dm + h * DH;
    const bf16_t* vtb = vt  + ((size_t)b * Dm + h * DH) * T;

    const int qrow0 = qt * 64 + wave * 16;

    v16bf qa[2];
#pragma unroll
    for (int s = 0; s < 2; ++s) qa[s] = load_a32(qb, Dm, qrow0, 32 * s, lane);

    v8f z = {0.f, 0.f, 0.f, 0.f, 0.f, 0.f, 0.f, 0.f};
    v8f oc[4];
#pragma unroll
    for (int t = 0; t < 4; ++t) oc[t] = z;
    float m[8], l[8];
#pragma unroll
    for (int j = 0; j < 8; ++j) { m[j] = -__builtin_inff(); l[j] = 0.f; }

    for (int kt = 0; kt <= qt; ++kt) {
        // ---- S = Q K^T ----
        v8f sc[4];
#pragma unroll
        for (int t = 0; t < 4; ++t) sc[t] = z;
#pragma unroll
        for (int s = 0; s < 2; ++s) {
#pragma unroll
            for (int t = 0; t < 4; ++t) {
                const v16bf* bp = reinterpret_cast<const v16bf*>(
                    kb + (size_t)(kt * 64 + 16 * t + cc) * Dm + 32 * s + 16 * half);
                sc[t] = __builtin_amdgcn_wmma_f32_16x16x32_bf16(
                            false, qa[s], false, *bp, (short)0, sc[t], false, false);
            }
        }

        // ---- scale + causal mask + online softmax ----
        const bool diag = (kt == qt);
        float sv[4][8];
#pragma unroll
        for (int j = 0; j < 8; ++j) {
            const int qrow = qrow0 + j + 8 * half;
            float mj = -__builtin_inff();
#pragma unroll
            for (int t = 0; t < 4; ++t) {
                int key = kt * 64 + 16 * t + cc;
                float s1 = sc[t][j] * 0.125f;
                if (diag && key > qrow) s1 = -__builtin_inff();
                sv[t][j] = s1;
                mj = fmaxf(mj, s1);
            }
#pragma unroll
            for (int o = 8; o > 0; o >>= 1) mj = fmaxf(mj, __shfl_xor(mj, o, 16));
            float mnew = fmaxf(m[j], mj);
            float f = __expf(m[j] - mnew);
            float psum = 0.f;
#pragma unroll
            for (int t = 0; t < 4; ++t) {
                float p = __expf(sv[t][j] - mnew);
                psum += p;
                myp[(j + 8 * half) * 64 + 16 * t + cc] = f2b(p);
            }
#pragma unroll
            for (int o = 8; o > 0; o >>= 1) psum += __shfl_xor(psum, o, 16);
            l[j] = l[j] * f + psum;
            m[j] = mnew;
#pragma unroll
            for (int t = 0; t < 4; ++t) oc[t][j] *= f;
        }

        // ---- O += P V  (P from LDS as A-frags; V^T contiguous B-frags) ----
#pragma unroll
        for (int s = 0; s < 2; ++s) {
            v16bf pa = load_a32(myp, 64, 0, 32 * s, lane);
#pragma unroll
            for (int t = 0; t < 4; ++t) {
                const v16bf* vp = reinterpret_cast<const v16bf*>(
                    vtb + (size_t)(16 * t + cc) * T + kt * 64 + 32 * s + 16 * half);
                oc[t] = __builtin_amdgcn_wmma_f32_16x16x32_bf16(
                            false, pa, false, *vp, (short)0, oc[t], false, false);
            }
        }
    }

    // ---- normalize + store (bf16, feeds proj GEMM) ----
#pragma unroll
    for (int t = 0; t < 4; ++t) {
#pragma unroll
        for (int j = 0; j < 8; ++j) {
            int row = qrow0 + j + 8 * half;
            float v = oc[t][j] / l[j];
            ybf[(size_t)(b * T + row) * Dm + h * DH + 16 * t + cc] = f2b(v);
        }
    }
}

// ---------------------------------------------------------------------------
// Host orchestration
// ---------------------------------------------------------------------------
extern "C" void kernel_launch(void* const* d_in, const int* in_sizes, int n_in,
                              void* d_out, int out_size, void* d_ws, size_t ws_size,
                              hipStream_t stream) {
    (void)in_sizes; (void)n_in; (void)out_size; (void)ws_size;
    const int B = 4, T = 1024, L = 6, D = 768, FULL = 622;
    const int M = B * T;                 // 4096
    const int D4 = 4 * D;                // 3072

    const int*   idxs  = (const int*)  d_in[0];
    const float* lat   = (const float*)d_in[1];
    const float* lon   = (const float*)d_in[2];
    const float* sog   = (const float*)d_in[3];
    const float* cog   = (const float*)d_in[4];
    const float* pos   = (const float*)d_in[5];
    const float* Wq    = (const float*)d_in[6];
    const float* bq    = (const float*)d_in[7];
    const float* Wk    = (const float*)d_in[8];
    const float* bk    = (const float*)d_in[9];
    const float* Wv    = (const float*)d_in[10];
    const float* bv    = (const float*)d_in[11];
    const float* Wp    = (const float*)d_in[12];
    const float* bp    = (const float*)d_in[13];
    const float* ln1g  = (const float*)d_in[14];
    const float* ln1b  = (const float*)d_in[15];
    const float* ln2g  = (const float*)d_in[16];
    const float* ln2b  = (const float*)d_in[17];
    const float* W1    = (const float*)d_in[18];
    const float* b1    = (const float*)d_in[19];
    const float* W2    = (const float*)d_in[20];
    const float* b2    = (const float*)d_in[21];
    const float* lnfg  = (const float*)d_in[22];
    const float* lnfb  = (const float*)d_in[23];
    const float* headw = (const float*)d_in[24];

    // workspace carve-out (256B aligned)
    char* ws = (char*)d_ws;
    size_t off = 0;
    auto alloc = [&](size_t bytes) -> void* {
        void* p = ws + off;
        off = (off + bytes + 255) & ~(size_t)255;
        return p;
    };
    float*  xf     = (float*) alloc((size_t)M * D * 4);
    bf16_t* hbf    = (bf16_t*)alloc((size_t)M * D * 2);
    bf16_t* qbf    = (bf16_t*)alloc((size_t)M * D * 2);
    bf16_t* kbf    = (bf16_t*)alloc((size_t)M * D * 2);
    bf16_t* vtbf   = (bf16_t*)alloc((size_t)M * D * 2);   // transposed V [B,H,DH,T]
    bf16_t* ybf    = (bf16_t*)alloc((size_t)M * D * 2);
    bf16_t* m1bf   = (bf16_t*)alloc((size_t)M * D4 * 2);
    bf16_t* wq_t   = (bf16_t*)alloc((size_t)L * D * D * 2);    // all weights [N,K]
    bf16_t* wk_t   = (bf16_t*)alloc((size_t)L * D * D * 2);
    bf16_t* wv_t   = (bf16_t*)alloc((size_t)L * D * D * 2);
    bf16_t* wp_t   = (bf16_t*)alloc((size_t)L * D * D * 2);
    bf16_t* w1_t   = (bf16_t*)alloc((size_t)L * D * D4 * 2);
    bf16_t* w2_t   = (bf16_t*)alloc((size_t)L * D4 * D * 2);
    bf16_t* head_t = (bf16_t*)alloc((size_t)FULL * D * 2);

    // f32 -> bf16 transposing conversions: in[K,N] per layer -> out[N,K]
    auto convT = [&](const float* src, bf16_t* dst, int K, int N, int layers) {
        dim3 g((N + 31) / 32, (K + 31) / 32, layers);
        f2bf_t_kernel<<<g, dim3(32, 8), 0, stream>>>(src, dst, K, N);
    };
    convT(Wq,    wq_t,   D,  D,    L);
    convT(Wk,    wk_t,   D,  D,    L);
    convT(Wv,    wv_t,   D,  D,    L);
    convT(Wp,    wp_t,   D,  D,    L);
    convT(W1,    w1_t,   D,  D4,   L);
    convT(W2,    w2_t,   D4, D,    L);
    convT(headw, head_t, D,  FULL, 1);

    embed_kernel<<<dim3((M * D + 255) / 256), 256, 0, stream>>>(
        idxs, lat, lon, sog, cog, pos, xf);

    const dim3 gq(M / 256, D  / 64);           // 16 x 12
    const dim3 g1(M / 256, D4 / 64);           // 16 x 48
    const dim3 gh(M / 256, (FULL + 63) / 64);  // 16 x 10
    const dim3 ga(B * 12, T / 64);             // 48 x 16

    for (int lyr = 0; lyr < L; ++lyr) {
        const size_t wo  = (size_t)lyr * D * D;
        const size_t wo1 = (size_t)lyr * D * D4;

        ln_kernel<<<M, 256, 0, stream>>>(xf, ln1g + lyr * D, ln1b + lyr * D, hbf);

        gemm_bf16_kernel<ACT_NONE, false, false, true, false><<<gq, 256, 0, stream>>>(
            hbf, wq_t + wo, bq + lyr * D, nullptr, nullptr, qbf, M, D, D);
        gemm_bf16_kernel<ACT_NONE, false, false, true, false><<<gq, 256, 0, stream>>>(
            hbf, wk_t + wo, bk + lyr * D, nullptr, nullptr, kbf, M, D, D);
        gemm_bf16_kernel<ACT_NONE, false, false, false, true><<<gq, 256, 0, stream>>>(
            hbf, wv_t + wo, bv + lyr * D, nullptr, nullptr, vtbf, M, D, D);

        attn_kernel<<<ga, 128, 0, stream>>>(qbf, kbf, vtbf, ybf);

        gemm_bf16_kernel<ACT_NONE, true, true, false, false><<<gq, 256, 0, stream>>>(
            ybf, wp_t + wo, bp + lyr * D, xf, xf, nullptr, M, D, D);

        ln_kernel<<<M, 256, 0, stream>>>(xf, ln2g + lyr * D, ln2b + lyr * D, hbf);

        gemm_bf16_kernel<ACT_GELU, false, false, true, false><<<g1, 256, 0, stream>>>(
            hbf, w1_t + wo1, b1 + lyr * D4, nullptr, nullptr, m1bf, M, D4, D);

        gemm_bf16_kernel<ACT_NONE, true, true, false, false><<<gq, 256, 0, stream>>>(
            m1bf, w2_t + wo1, b2 + lyr * D, xf, xf, nullptr, M, D, D4);
    }

    ln_kernel<<<M, 256, 0, stream>>>(xf, lnfg, lnfb, hbf);

    gemm_bf16_kernel<ACT_NONE, false, true, false, false><<<gh, 256, 0, stream>>>(
        hbf, head_t, nullptr, nullptr, (float*)d_out, nullptr, M, FULL, D);
}
